// mHCGenerator_80951543595207
// MI455X (gfx1250) — matrix operand order repaired and checked by reading, compiled, not run on hardware
//
#include <hip/hip_runtime.h>
#include <hip/hip_bf16.h>
#include <math.h>

typedef __attribute__((ext_vector_type(16))) __bf16 v16bf;
typedef __attribute__((ext_vector_type(8)))  float  v8f;

#define IN_CH   768
#define HWSZ    196
#define BATCH   128
#define RMS_EPS 1.1920929e-07f
#define SK_EPS  1e-6f

// -------- Kernel 1: global average pool over H*W (bandwidth bound) --------
// One wave32 per (b,c) plane of 196 contiguous floats (49 float4's).
__global__ __launch_bounds__(256) void gap_kernel(const float* __restrict__ x,
                                                  float* __restrict__ xgap) {
    const int wave = threadIdx.x >> 5;
    const int lane = threadIdx.x & 31;
    const long plane = (long)blockIdx.x * 8 + wave;      // 98304 planes total
    const float4* p = (const float4*)(x + plane * HWSZ);
    float4 v = p[lane];
    float s = v.x + v.y + v.z + v.w;
    if (lane < 17) {                                     // indices 32..48
        float4 v2 = p[lane + 32];
        s += v2.x + v2.y + v2.z + v2.w;
    }
    #pragma unroll
    for (int m = 16; m >= 1; m >>= 1) s += __shfl_xor(s, m, 32);
    if (lane == 0) xgap[plane] = s * (1.0f / (float)HWSZ);
}

// -------- Kernel 2: RMSNorm + [16,768]x[768,32] GEMM tile via WMMA bf16 ----
// One wave per 16-row tile of the batch (grid = 8). N=0..23 valid, padded to 32.
__global__ __launch_bounds__(32) void gemm_kernel(const float* __restrict__ xgap,
                                                  const float* __restrict__ Wpre,
                                                  const float* __restrict__ Wpost,
                                                  const float* __restrict__ Wres,
                                                  const float* __restrict__ normw,
                                                  float* __restrict__ lin) {
    __shared__ __bf16 xn[16][IN_CH];
    const int lane = threadIdx.x;          // 0..31
    const int m0   = blockIdx.x * 16;
    const int row  = lane & 15;
    const int half = lane >> 4;

    // pass 1: per-row sum of squares (2 lanes per row)
    const float* xr = xgap + (long)(m0 + row) * IN_CH + half * (IN_CH / 2);
    float sq = 0.0f;
    for (int c = 0; c < IN_CH / 2; ++c) { float v = xr[c]; sq += v * v; }
    sq += __shfl_xor(sq, 16, 32);
    const float inv = 1.0f / sqrtf(sq * (1.0f / (float)IN_CH) + RMS_EPS);

    // pass 2: normalize * norm_w, stage bf16 tile in LDS
    const float* nwp = normw + half * (IN_CH / 2);
    for (int c = 0; c < IN_CH / 2; ++c)
        xn[row][half * (IN_CH / 2) + c] = (__bf16)(xr[c] * inv * nwp[c]);
    __syncthreads();

    v8f acc0 = {};   // N = 0..15
    v8f acc1 = {};   // N = 16..31 (16..23 valid)
    for (int k0 = 0; k0 < IN_CH; k0 += 32) {
        // A fragment: 16x32 bf16, lane = M, dword v holds K pair
        v16bf a;
        #pragma unroll
        for (int v = 0; v < 8; ++v) {
            const int kb = ((v >> 2) << 4) + (half << 3) + ((v & 3) << 1);
            a[2 * v]     = xn[row][k0 + kb];
            a[2 * v + 1] = xn[row][k0 + kb + 1];
        }
        // B fragments: 32x16 bf16, lane%16 = N, element = K (same K order as A); B[k][n] = W_all[n][k]
        const float* wr = (row < 4) ? (Wpre + row * IN_CH)
                        : (row < 8) ? (Wpost + (row - 4) * IN_CH)
                                    : (Wres + (row - 8) * IN_CH);
        v16bf b0, b1;
        #pragma unroll
        for (int e = 0; e < 16; ++e) {
            const int k = k0 + ((e >> 3) << 4) + (half << 3) + (e & 7);
            b0[e] = (__bf16)wr[k];
            b1[e] = (row < 8) ? (__bf16)Wres[(8 + row) * IN_CH + k] : (__bf16)0.0f;
        }

        acc0 = __builtin_amdgcn_wmma_f32_16x16x32_bf16(false, a, false, b0,
                                                       (short)0, acc0, false, false);
        acc1 = __builtin_amdgcn_wmma_f32_16x16x32_bf16(false, a, false, b1,
                                                       (short)0, acc1, false, false);
    }

    // D layout: element r, lane L -> row r+8*(L/16), col L%16
    #pragma unroll
    for (int r = 0; r < 8; ++r) {
        const int m = m0 + r + 8 * half;
        lin[m * 32 + row]      = acc0[r];
        lin[m * 32 + 16 + row] = acc1[r];
    }
}

// -------- Kernel 3: gates + per-row 4x4 Sinkhorn-Knopp (registers) --------
__global__ __launch_bounds__(128) void head_kernel(const float* __restrict__ lin,
                                                   const float* __restrict__ bpre,
                                                   const float* __restrict__ bpost,
                                                   const float* __restrict__ bres,
                                                   const float* __restrict__ apre_p,
                                                   const float* __restrict__ apost_p,
                                                   const float* __restrict__ ares_p,
                                                   float* __restrict__ out) {
    const int b = threadIdx.x;                 // 0..127 (one block)
    const float apre = apre_p[0], apost = apost_p[0], ares = ares_p[0];
    const float* v = lin + b * 32;
    float* Hpre  = out;                        // [128,4]
    float* Hpost = out + BATCH * 4;            // [128,4]
    float* Hres  = out + BATCH * 8;            // [128,16]

    #pragma unroll
    for (int n = 0; n < 4; ++n) {
        const float p = apre * v[n] + bpre[n];
        Hpre[b * 4 + n] = 1.0f / (1.0f + expf(-p));
        const float q = apost * v[4 + n] + bpost[n];
        Hpost[b * 4 + n] = 2.0f / (1.0f + expf(-q));
    }

    float M[4][4];
    #pragma unroll
    for (int i = 0; i < 4; ++i) {
        float r[4];
        float mx = -INFINITY;
        #pragma unroll
        for (int j = 0; j < 4; ++j) {
            r[j] = ares * v[8 + i * 4 + j] + bres[i * 4 + j];
            mx = fmaxf(mx, r[j]);
        }
        #pragma unroll
        for (int j = 0; j < 4; ++j) M[i][j] = expf(r[j] - mx);
    }
    for (int it = 0; it < 20; ++it) {
        #pragma unroll
        for (int i = 0; i < 4; ++i) {
            const float s  = M[i][0] + M[i][1] + M[i][2] + M[i][3];
            const float is = 1.0f / (s + SK_EPS);
            #pragma unroll
            for (int j = 0; j < 4; ++j) M[i][j] *= is;
        }
        #pragma unroll
        for (int j = 0; j < 4; ++j) {
            const float s  = M[0][j] + M[1][j] + M[2][j] + M[3][j];
            const float is = 1.0f / (s + SK_EPS);
            #pragma unroll
            for (int i = 0; i < 4; ++i) M[i][j] *= is;
        }
    }
    #pragma unroll
    for (int i = 0; i < 4; ++i)
        #pragma unroll
        for (int j = 0; j < 4; ++j)
            Hres[b * 16 + i * 4 + j] = M[i][j];
}

extern "C" void kernel_launch(void* const* d_in, const int* in_sizes, int n_in,
                              void* d_out, int out_size, void* d_ws, size_t ws_size,
                              hipStream_t stream) {
    (void)in_sizes; (void)n_in; (void)out_size; (void)ws_size;
    const float* x     = (const float*)d_in[0];   // [128,768,14,14]
    const float* Wpre  = (const float*)d_in[1];   // [4,768]
    const float* Wpost = (const float*)d_in[2];   // [4,768]
    const float* Wres  = (const float*)d_in[3];   // [16,768]
    const float* bpre  = (const float*)d_in[4];   // [4]
    const float* bpost = (const float*)d_in[5];   // [4]
    const float* bres  = (const float*)d_in[6];   // [16]
    const float* apre  = (const float*)d_in[7];   // scalar
    const float* apost = (const float*)d_in[8];   // scalar
    const float* ares  = (const float*)d_in[9];   // scalar
    const float* normw = (const float*)d_in[10];  // [768]

    float* xgap = (float*)d_ws;                          // 128*768 floats (384 KB)
    float* lin  = (float*)d_ws + BATCH * IN_CH;          // 128*32 floats (16 KB)
    float* outp = (float*)d_out;

    gap_kernel<<<(BATCH * IN_CH) / 8, 256, 0, stream>>>(x, xgap);
    gemm_kernel<<<BATCH / 16, 32, 0, stream>>>(xgap, Wpre, Wpost, Wres, normw, lin);
    head_kernel<<<1, BATCH, 0, stream>>>(lin, bpre, bpost, bres, apre, apost, ares, outp);
}
